// MyTransformer_31044023616460
// MI455X (gfx1250) — matrix operand order repaired and checked by reading, not compiled, not run on hardware
//
#include <hip/hip_runtime.h>

typedef __attribute__((ext_vector_type(16))) __bf16       v16bf;
typedef __attribute__((ext_vector_type(8)))  float        v8f;
typedef __attribute__((ext_vector_type(8)))  unsigned int v8u;

// ---------------------------------------------------------------------------
// Helpers
// ---------------------------------------------------------------------------
#if __has_builtin(__builtin_amdgcn_cvt_pk_bf16_f32)
__device__ __forceinline__ unsigned int pack_bf16(float a, float b) {
  auto r = __builtin_amdgcn_cvt_pk_bf16_f32(a, b);
  return __builtin_bit_cast(unsigned int, r);
}
#else
// Single-instruction packed f32->bf16 (RNE). Builtin not exposed by clang for
// gfx1250, but the hardware instruction exists (bf16 is first-class on CDNA5).
__device__ __forceinline__ unsigned int pack_bf16(float a, float b) {
  unsigned int r;
  asm("v_cvt_pk_bf16_f32 %0, %1, %2" : "=v"(r) : "v"(a), "v"(b));
  return r;
}
#endif

__device__ __forceinline__ v8f vzero8() {
  v8f z;
#pragma unroll
  for (int i = 0; i < 8; ++i) z[i] = 0.0f;
  return z;
}

// A-operand fragment (16xK tile, 16-bit elems, K-step 32), per CDNA5 ISA:
// lanes 0-15: M=lane, K pairs {0,2,4,6,16,18,20,22}; lanes 16-31: M=lane-16, +8.
// LDS layout: [row][k], k contiguous, ld in elements (even).
__device__ __forceinline__ void load_frag_a(const unsigned short* base, int ld,
                                            int lane, v8u& f) {
  int r  = lane & 15;
  int kb = (lane & 16) ? 8 : 0;
  const unsigned short* p = base + r * ld;
#pragma unroll
  for (int v = 0; v < 8; ++v) {
    int k = kb + 2 * (v & 3) + ((v & 4) << 2);
    f[v] = *(const unsigned int*)(p + k);
  }
}

// B-operand fragment (Kx16 tile): lanes 0-15: N=lane, K pairs {0,2,4,6,16,18,20,22};
// lanes 16-31: N=lane-16, +8. LDS layout: B transposed [n][k].
__device__ __forceinline__ void load_frag_b(const unsigned short* base, int ld,
                                            int lane, v8u& f) {
  int n  = lane & 15;
  int kb = (lane & 16) ? 8 : 0;
  const unsigned short* p = base + n * ld + kb;
#pragma unroll
  for (int v = 0; v < 8; ++v) f[v] = *(const unsigned int*)(p + 2 * (v & 3) + ((v & 4) << 2));
}

#define WMMA_BF16(A, B, C)                                                     \
  __builtin_amdgcn_wmma_f32_16x16x32_bf16(                                     \
      false, __builtin_bit_cast(v16bf, A), false, __builtin_bit_cast(v16bf, B),\
      (short)0, (C), false, false)

// ---------------------------------------------------------------------------
// GEMM: C[M,N] = A[M,K] * B[K,N] + bias (+optional ReLU). fp32 in/out, bf16 MACs.
// Block tile 128x128, 256 threads = 8 waves (4 M x 2 N), each wave 32x64.
// Double-buffered LDS; next K-tile staged via registers so global loads issue
// before the WMMA section (latency hidden under matrix math).
// ---------------------------------------------------------------------------
__launch_bounds__(256)
__global__ void gemm_bf16w(const float* __restrict__ A, const float* __restrict__ B,
                           const float* __restrict__ bias, float* __restrict__ C,
                           int M, int N, int K, int relu) {
  __shared__ __align__(16) unsigned short As[2][128 * 32];  // [m][k]
  __shared__ __align__(16) unsigned short Bs[2][128 * 32];  // [n][k] (B^T)

  int t = threadIdx.x, lane = t & 31, wave = t >> 5;
  int waveM = wave >> 1, waveN = wave & 1;
  int m0 = blockIdx.y * 128, n0 = blockIdx.x * 128;

  v8f acc[2][4];
#pragma unroll
  for (int i = 0; i < 2; ++i)
#pragma unroll
    for (int j = 0; j < 4; ++j) acc[i][j] = vzero8();

  float4 ar[4];          // A stage: 4 x float4 = 16 elems/thread
  float2 br0[4], br1[4]; // B stage: 4 x (2x2 block) = 16 elems/thread

  auto loadAB = [&](int k0) {
#pragma unroll
    for (int i = 0; i < 4; ++i) {            // A tile 128x32: f4 id -> (m, k/4)
      int f = t + 256 * i;
      int m = f >> 3, k = (f & 7) << 2;
      ar[i] = *(const float4*)&A[(size_t)(m0 + m) * K + (k0 + k)];
    }
#pragma unroll
    for (int i = 0; i < 4; ++i) {            // B tile 32x128: unit -> (k2, n2)
      int u = t + 256 * i;
      int n = (u & 63) << 1, k = (u >> 6) << 1;
      br0[i] = *(const float2*)&B[(size_t)(k0 + k) * N + (n0 + n)];
      br1[i] = *(const float2*)&B[(size_t)(k0 + k + 1) * N + (n0 + n)];
    }
  };
  auto storeAB = [&](int buf) {
#pragma unroll
    for (int i = 0; i < 4; ++i) {
      int f = t + 256 * i;
      int m = f >> 3, k = (f & 7) << 2;
      uint2 w;
      w.x = pack_bf16(ar[i].x, ar[i].y);
      w.y = pack_bf16(ar[i].z, ar[i].w);
      *(uint2*)&As[buf][m * 32 + k] = w;
    }
#pragma unroll
    for (int i = 0; i < 4; ++i) {
      int u = t + 256 * i;
      int n = (u & 63) << 1, k = (u >> 6) << 1;
      *(unsigned int*)&Bs[buf][n * 32 + k]       = pack_bf16(br0[i].x, br1[i].x);
      *(unsigned int*)&Bs[buf][(n + 1) * 32 + k] = pack_bf16(br0[i].y, br1[i].y);
    }
  };

  int nk = K >> 5;
  loadAB(0);
  storeAB(0);
  __syncthreads();

  for (int kt = 0; kt < nk; ++kt) {
    int cur = kt & 1;
    if (kt + 1 < nk) loadAB((kt + 1) << 5);   // issue next tile's global loads

    v8u af[2], bfr[4];
#pragma unroll
    for (int mt = 0; mt < 2; ++mt)
      load_frag_a(&As[cur][(waveM * 32 + mt * 16) * 32], 32, lane, af[mt]);
#pragma unroll
    for (int nt = 0; nt < 4; ++nt)
      load_frag_b(&Bs[cur][(waveN * 64 + nt * 16) * 32], 32, lane, bfr[nt]);
#pragma unroll
    for (int mt = 0; mt < 2; ++mt)
#pragma unroll
      for (int nt = 0; nt < 4; ++nt)
        acc[mt][nt] = WMMA_BF16(af[mt], bfr[nt], acc[mt][nt]);

    if (kt + 1 < nk) storeAB((kt + 1) & 1);   // convert + LDS store after math
    __syncthreads();
  }

  int mo = (lane & 16) ? 8 : 0;
  int nn = lane & 15;
#pragma unroll
  for (int mt = 0; mt < 2; ++mt)
#pragma unroll
    for (int nt = 0; nt < 4; ++nt) {
      int n = n0 + waveN * 64 + nt * 16 + nn;
      float bb = bias ? bias[n] : 0.0f;
#pragma unroll
      for (int r = 0; r < 8; ++r) {
        int m = m0 + waveM * 32 + mt * 16 + mo + r;
        float v = acc[mt][nt][r] + bb;
        if (relu) v = fmaxf(v, 0.0f);
        C[(size_t)m * N + n] = v;
      }
    }
}

// ---------------------------------------------------------------------------
// Flash attention: per (b, h, 128 q-rows) WG; 8 waves each own 16 q-rows.
// KV streamed in 32-row blocks, double-buffered; online softmax; WMMA for
// QK^T and P*V. Softmax scale (1/sqrt(64)) folded into Q at staging.
// ---------------------------------------------------------------------------
__launch_bounds__(256)
__global__ void flash_attn(const float* __restrict__ Q, const float* __restrict__ Kg,
                           const float* __restrict__ Vg, float* __restrict__ O,
                           int S, int D) {
  __shared__ __align__(16) unsigned short Qs[128 * 64];      // [m][d]
  __shared__ __align__(16) unsigned short Ks[2][32 * 64];    // [kv][d]
  __shared__ __align__(16) unsigned short Vt[2][64 * 32];    // [d][kv]
  __shared__ __align__(16) unsigned short Ps[8 * 16 * 32];   // per-wave P [m][kv]

  int t = threadIdx.x, lane = t & 31, wave = t >> 5;
  int b = blockIdx.z, h = blockIdx.y;
  int q0 = blockIdx.x * 128;
  const float* Qb = Q + (size_t)b * S * D + (size_t)h * 64;
  const float* Kb = Kg + (size_t)b * S * D + (size_t)h * 64;
  const float* Vb = Vg + (size_t)b * S * D + (size_t)h * 64;
  float*       Ob = O + (size_t)b * S * D + (size_t)h * 64;

#pragma unroll
  for (int i = 0; i < 8; ++i) {             // stage Q 128x64 (scaled), float4
    int f = t + 256 * i;
    int m = f >> 4, d = (f & 15) << 2;
    float4 q4 = *(const float4*)&Qb[(size_t)(q0 + m) * D + d];
    uint2 w;
    w.x = pack_bf16(q4.x * 0.125f, q4.y * 0.125f);
    w.y = pack_bf16(q4.z * 0.125f, q4.w * 0.125f);
    *(uint2*)&Qs[m * 64 + d] = w;
  }
  __syncthreads();

  v8u qf[2];
  load_frag_a(&Qs[(wave * 16) * 64], 64, lane, qf[0]);
  load_frag_a(&Qs[(wave * 16) * 64 + 32], 64, lane, qf[1]);

  v8f oacc[4];
#pragma unroll
  for (int d = 0; d < 4; ++d) oacc[d] = vzero8();
  float mrow[8], lrow[8];
#pragma unroll
  for (int r = 0; r < 8; ++r) { mrow[r] = -3.0e38f; lrow[r] = 0.0f; }

  int mo = (lane & 16) ? 8 : 0;
  int nn = lane & 15;
  unsigned short* Pw = &Ps[wave * 16 * 32];

  float4 kr[2];
  float2 vr0[2], vr1[2];
  auto loadKV = [&](int kv0) {
#pragma unroll
    for (int i = 0; i < 2; ++i) {           // K block 32x64, float4
      int f = t + 256 * i;
      int m = f >> 4, d = (f & 15) << 2;
      kr[i] = *(const float4*)&Kb[(size_t)(kv0 + m) * D + d];
    }
#pragma unroll
    for (int i = 0; i < 2; ++i) {           // V block 32x64, 2x2 units
      int u = t + 256 * i;
      int d = (u & 31) << 1, kv = (u >> 5) << 1;
      vr0[i] = *(const float2*)&Vb[(size_t)(kv0 + kv) * D + d];
      vr1[i] = *(const float2*)&Vb[(size_t)(kv0 + kv + 1) * D + d];
    }
  };
  auto storeKV = [&](int buf) {
#pragma unroll
    for (int i = 0; i < 2; ++i) {
      int f = t + 256 * i;
      int m = f >> 4, d = (f & 15) << 2;
      uint2 w;
      w.x = pack_bf16(kr[i].x, kr[i].y);
      w.y = pack_bf16(kr[i].z, kr[i].w);
      *(uint2*)&Ks[buf][m * 64 + d] = w;
    }
#pragma unroll
    for (int i = 0; i < 2; ++i) {           // V transposed -> [d][kv]
      int u = t + 256 * i;
      int d = (u & 31) << 1, kv = (u >> 5) << 1;
      *(unsigned int*)&Vt[buf][d * 32 + kv]       = pack_bf16(vr0[i].x, vr1[i].x);
      *(unsigned int*)&Vt[buf][(d + 1) * 32 + kv] = pack_bf16(vr0[i].y, vr1[i].y);
    }
  };

  int nblk = S >> 5;
  loadKV(0);
  storeKV(0);
  __syncthreads();

  for (int j = 0; j < nblk; ++j) {
    int cur = j & 1;
    if (j + 1 < nblk) loadKV((j + 1) << 5);  // issue next block's loads early

    // S tile 16x32 = Q(16x64) x K^T : two 16x16 halves, K-dim split 2x32
    v8f sc[2];
#pragma unroll
    for (int th = 0; th < 2; ++th) {
      sc[th] = vzero8();
#pragma unroll
      for (int kk = 0; kk < 2; ++kk) {
        v8u kf;
        load_frag_b(&Ks[cur][(th * 16) * 64 + kk * 32], 64, lane, kf);
        sc[th] = WMMA_BF16(qf[kk], kf, sc[th]);
      }
    }

    // Online softmax update (row m = r + mo, stats replicated over 16 lanes)
#pragma unroll
    for (int r = 0; r < 8; ++r) {
      float s0 = sc[0][r], s1 = sc[1][r];
      float mx = fmaxf(s0, s1);
#pragma unroll
      for (int off = 1; off < 16; off <<= 1) mx = fmaxf(mx, __shfl_xor(mx, off, 32));
      float mnew = fmaxf(mrow[r], mx);
      float p0 = __expf(s0 - mnew);
      float p1 = __expf(s1 - mnew);
      float ps = p0 + p1;
#pragma unroll
      for (int off = 1; off < 16; off <<= 1) ps += __shfl_xor(ps, off, 32);
      float corr = __expf(mrow[r] - mnew);
      lrow[r] = lrow[r] * corr + ps;
      mrow[r] = mnew;
#pragma unroll
      for (int d = 0; d < 4; ++d) oacc[d][r] *= corr;
      int m = r + mo;
      unsigned int pk = pack_bf16(p0, p1);   // one cvt; lo/hi 16-bit DS stores
      Pw[m * 32 + nn]      = (unsigned short)pk;
      Pw[m * 32 + 16 + nn] = (unsigned short)(pk >> 16);
    }

    // O += P(16x32) x V(32x64)
    v8u pf;
    load_frag_a(Pw, 32, lane, pf);
#pragma unroll
    for (int dt = 0; dt < 4; ++dt) {
      v8u vf;
      load_frag_b(&Vt[cur][(dt * 16) * 32], 32, lane, vf);
      oacc[dt] = WMMA_BF16(pf, vf, oacc[dt]);
    }

    if (j + 1 < nblk) storeKV((j + 1) & 1);
    __syncthreads();
  }

#pragma unroll
  for (int r = 0; r < 8; ++r) {
    float inv = 1.0f / lrow[r];
    int m = q0 + wave * 16 + mo + r;
#pragma unroll
    for (int dt = 0; dt < 4; ++dt)
      Ob[(size_t)m * D + dt * 16 + nn] = oacc[dt][r] * inv;
  }
}

// ---------------------------------------------------------------------------
// Residual add + LayerNorm: one block per row, 256 threads x one float4 each.
// ---------------------------------------------------------------------------
__launch_bounds__(256)
__global__ void add_ln(const float* __restrict__ X, const float* __restrict__ R,
                       const float* __restrict__ gamma, const float* __restrict__ beta,
                       float* __restrict__ Out, int D) {
  __shared__ float red[8];
  size_t row = blockIdx.x;
  int t = threadIdx.x;
  int c = t << 2;
  float4 xv = *(const float4*)&X[row * D + c];
  float4 rv = *(const float4*)&R[row * D + c];
  float v[4] = {xv.x + rv.x, xv.y + rv.y, xv.z + rv.z, xv.w + rv.w};
  float sum = v[0] + v[1] + v[2] + v[3];
#pragma unroll
  for (int off = 16; off >= 1; off >>= 1) sum += __shfl_xor(sum, off, 32);
  if ((t & 31) == 0) red[t >> 5] = sum;
  __syncthreads();
  float tot = 0.0f;
#pragma unroll
  for (int i = 0; i < 8; ++i) tot += red[i];
  float mu = tot / (float)D;
  float var = 0.0f;
#pragma unroll
  for (int i = 0; i < 4; ++i) { float d = v[i] - mu; var += d * d; }
#pragma unroll
  for (int off = 16; off >= 1; off >>= 1) var += __shfl_xor(var, off, 32);
  __syncthreads();
  if ((t & 31) == 0) red[t >> 5] = var;
  __syncthreads();
  float tv = 0.0f;
#pragma unroll
  for (int i = 0; i < 8; ++i) tv += red[i];
  float rstd = rsqrtf(tv / (float)D + 1e-5f);
  float4 g = *(const float4*)&gamma[c];
  float4 be = *(const float4*)&beta[c];
  float4 o;
  o.x = (v[0] - mu) * rstd * g.x + be.x;
  o.y = (v[1] - mu) * rstd * g.y + be.y;
  o.z = (v[2] - mu) * rstd * g.z + be.z;
  o.w = (v[3] - mu) * rstd * g.w + be.w;
  *(float4*)&Out[row * D + c] = o;
}

// pe row indexed by *batch* (faithful to reference's pe[:B] broadcast over seq)
__global__ void posenc_add(float* __restrict__ Hb, int B, int S, int D) {
  int idx = blockIdx.x * blockDim.x + threadIdx.x;
  if (idx >= B * S * D) return;
  int d = idx % D;
  int b = (idx / D) / S;
  float div = __expf((float)(d & ~1) * (-9.210340371976184f / (float)D));
  float ang = (float)b * div;
  Hb[idx] += (d & 1) ? __cosf(ang) : __sinf(ang);
}

__global__ void pool_mean(const float* __restrict__ Hb, float* __restrict__ P,
                          int S, int D) {
  int idx = blockIdx.x * blockDim.x + threadIdx.x;  // one (b,d) per thread
  int b = idx / D, d = idx % D;
  const float* p = Hb + (size_t)b * S * D + d;
  float s = 0.0f;
  for (int i = 0; i < S; ++i) s += p[(size_t)i * D];
  P[idx] = s / (float)S;
}

__global__ void fc_out(const float* __restrict__ P, const float* __restrict__ W,
                       const float* __restrict__ bias, float* __restrict__ Out,
                       int D, int C) {
  int b = blockIdx.x;
  int wave = threadIdx.x >> 5, lane = threadIdx.x & 31;
  if (wave >= C) return;
  float s = 0.0f;
  for (int i = lane; i < D; i += 32) s += P[b * D + i] * W[(size_t)i * C + wave];
#pragma unroll
  for (int off = 16; off >= 1; off >>= 1) s += __shfl_xor(s, off, 32);
  if (lane == 0) Out[b * C + wave] = s + bias[wave];
}

// ---------------------------------------------------------------------------
// Orchestration
// ---------------------------------------------------------------------------
extern "C" void kernel_launch(void* const* d_in, const int* in_sizes, int n_in,
                              void* d_out, int out_size, void* d_ws, size_t ws_size,
                              hipStream_t stream) {
  (void)in_sizes; (void)n_in; (void)out_size; (void)ws_size;
  const int B = 8, S = 1024, IN = 256, D = 1024, Hh = 16, L = 6, C = 10;
  const int M = B * S;  // 8192

  const float* x     = (const float*)d_in[0];
  const float* emb_w = (const float*)d_in[1];
  const float* emb_b = (const float*)d_in[2];
  const float* Wq = (const float*)d_in[3];  const float* bq = (const float*)d_in[4];
  const float* Wk = (const float*)d_in[5];  const float* bk = (const float*)d_in[6];
  const float* Wv = (const float*)d_in[7];  const float* bv = (const float*)d_in[8];
  const float* Wo = (const float*)d_in[9];  const float* bo = (const float*)d_in[10];
  const float* W1 = (const float*)d_in[11]; const float* b1 = (const float*)d_in[12];
  const float* W2 = (const float*)d_in[13]; const float* b2 = (const float*)d_in[14];
  const float* ln1_s = (const float*)d_in[15]; const float* ln1_b = (const float*)d_in[16];
  const float* ln2_s = (const float*)d_in[17]; const float* ln2_b = (const float*)d_in[18];
  const float* fc_w = (const float*)d_in[19];  const float* fc_b = (const float*)d_in[20];

  float* ws = (float*)d_ws;
  size_t off = 0;
  float* hbuf = ws + off;  off += (size_t)M * D;
  float* qb   = ws + off;  off += (size_t)M * D;
  float* kb   = ws + off;  off += (size_t)M * D;
  float* vb   = ws + off;  off += (size_t)M * D;
  float* t1   = ws + off;  off += (size_t)M * 4 * D;
  float* t2   = ws + off;  off += (size_t)M * D;
  float* pooled = ws + off;

  dim3 blk(256);

  // Embedding + positional encoding
  gemm_bf16w<<<dim3(D / 128, M / 128), blk, 0, stream>>>(x, emb_w, emb_b, hbuf, M, D, IN, 0);
  posenc_add<<<(M * D + 255) / 256, blk, 0, stream>>>(hbuf, B, S, D);

  for (int l = 0; l < L; ++l) {
    const float* wq = Wq + (size_t)l * D * D;     const float* bql = bq + (size_t)l * D;
    const float* wk = Wk + (size_t)l * D * D;     const float* bkl = bk + (size_t)l * D;
    const float* wv = Wv + (size_t)l * D * D;     const float* bvl = bv + (size_t)l * D;
    const float* wo = Wo + (size_t)l * D * D;     const float* bol = bo + (size_t)l * D;
    const float* w1 = W1 + (size_t)l * D * 4 * D; const float* b1l = b1 + (size_t)l * 4 * D;
    const float* w2 = W2 + (size_t)l * 4 * D * D; const float* b2l = b2 + (size_t)l * D;

    gemm_bf16w<<<dim3(D / 128, M / 128), blk, 0, stream>>>(hbuf, wq, bql, qb, M, D, D, 0);
    gemm_bf16w<<<dim3(D / 128, M / 128), blk, 0, stream>>>(hbuf, wk, bkl, kb, M, D, D, 0);
    gemm_bf16w<<<dim3(D / 128, M / 128), blk, 0, stream>>>(hbuf, wv, bvl, vb, M, D, D, 0);

    flash_attn<<<dim3(S / 128, Hh, B), blk, 0, stream>>>(qb, kb, vb, t2, S, D);

    gemm_bf16w<<<dim3(D / 128, M / 128), blk, 0, stream>>>(t2, wo, bol, qb, M, D, D, 0);
    add_ln<<<dim3(M), blk, 0, stream>>>(hbuf, qb, ln1_s + (size_t)l * D, ln1_b + (size_t)l * D, hbuf, D);

    gemm_bf16w<<<dim3(4 * D / 128, M / 128), blk, 0, stream>>>(hbuf, w1, b1l, t1, M, 4 * D, D, 1);
    gemm_bf16w<<<dim3(D / 128, M / 128), blk, 0, stream>>>(t1, w2, b2l, t2, M, D, 4 * D, 0);
    add_ln<<<dim3(M), blk, 0, stream>>>(hbuf, t2, ln2_s + (size_t)l * D, ln2_b + (size_t)l * D, hbuf, D);
  }

  pool_mean<<<dim3((B * D) / 256), blk, 0, stream>>>(hbuf, pooled, S, D);
  fc_out<<<dim3(B), dim3(320), 0, stream>>>(pooled, fc_w, fc_b, (float*)d_out, D, C);
}
